// MultiTask_20512763806195
// MI455X (gfx1250) — matrix operand order repaired and checked by reading, not compiled, not run on hardware
//
#include <hip/hip_runtime.h>
#include <hip/hip_bf16.h>

// ---- CDNA5 WMMA vector types -------------------------------------------------
typedef __attribute__((ext_vector_type(16))) __bf16 v16bf;
typedef __attribute__((ext_vector_type(8)))  __bf16 v8bf;
typedef __attribute__((ext_vector_type(8)))  float  v8f;

union BF16x16 { v16bf v; v8bf h[2]; };

// round-to-nearest-even f32 -> bf16 (raw bits)
__device__ __forceinline__ unsigned int f2bf(float f) {
    unsigned int u = __float_as_uint(f);
    u += 0x7FFFu + ((u >> 16) & 1u);
    return u >> 16;
}
__device__ __forceinline__ unsigned int pack2(float lo, float hi) {
    return f2bf(lo) | (f2bf(hi) << 16);
}

// Problem constants
#define BATCH   16384
#define IDIM    512
#define UNITS   256
#define NEXP    8
#define NTASK   4
#define NCOLS   (UNITS * NEXP)   // 2048

// Tiling
#define BM 128
#define BN 128
#define BK 32
#define LDEO 132                 // floats per Eo row (128 + pad)

#define SMEM_GATE_BYTES (BM * 32 * 4)                     // 16384
#define SMEM_EO_BYTES   (BM * LDEO * 4)                   // 67584
#define SMEM_TOTAL      (SMEM_GATE_BYTES + SMEM_EO_BYTES) // 83968

// workspace layout (bytes)
#define WS_GATES_OFF 0
#define WS_XB_OFF    (BATCH * 32 * 4)                         // 2 MB
#define WS_WET_OFF   (WS_XB_OFF + BATCH * IDIM * 2)           // +16 MB
// total = 20 MB

// ---- Prep 1: x f32 -> bf16, same [B][K] layout ------------------------------
__global__ __launch_bounds__(256) void convx_kernel(
    const float* __restrict__ x, unsigned short* __restrict__ xb) {
    size_t base = ((size_t)blockIdx.x * 256 + threadIdx.x) * 8;
    float4 v0 = *(const float4*)(x + base);
    float4 v1 = *(const float4*)(x + base + 4);
    uint4 u;
    u.x = pack2(v0.x, v0.y); u.y = pack2(v0.z, v0.w);
    u.z = pack2(v1.x, v1.y); u.w = pack2(v1.z, v1.w);
    *(uint4*)(xb + base) = u;
}

// ---- Prep 2: We f32 [K][N] -> bf16 transposed [N][K] ------------------------
#define TT 64
#define TLD 72   // LDS row stride in halfs (144 B, 16B-aligned)
__global__ __launch_bounds__(256) void wet_kernel(
    const float* __restrict__ We, unsigned short* __restrict__ wet) {
    __shared__ unsigned short tile[TT * TLD];
    const int tid = threadIdx.x;
    const int k0 = blockIdx.x * TT;   // 8 blocks
    const int n0 = blockIdx.y * TT;   // 32 blocks
    // read coalesced along n, scatter into [n][k] tile
#pragma unroll
    for (int p = 0; p < 4; ++p) {
        int lin = tid + p * 256;
        int k = lin >> 4, nq = lin & 15;
        float4 v = *(const float4*)(We + (size_t)(k0 + k) * NCOLS + n0 + nq * 4);
        tile[(nq * 4 + 0) * TLD + k] = (unsigned short)f2bf(v.x);
        tile[(nq * 4 + 1) * TLD + k] = (unsigned short)f2bf(v.y);
        tile[(nq * 4 + 2) * TLD + k] = (unsigned short)f2bf(v.z);
        tile[(nq * 4 + 3) * TLD + k] = (unsigned short)f2bf(v.w);
    }
    __syncthreads();
    // write coalesced along k
    {
        int n = tid >> 2, kc = tid & 3;
        uint4 a = *(const uint4*)&tile[n * TLD + kc * 16];
        uint4 b = *(const uint4*)&tile[n * TLD + kc * 16 + 8];
        unsigned short* dst = wet + (size_t)(n0 + n) * IDIM + k0 + kc * 16;
        *(uint4*)dst = a;
        *(uint4*)(dst + 8) = b;
    }
}

// ---- Kernel 3: gate logits + softmax over experts ---------------------------
// gates layout: [b][t][e]
__global__ __launch_bounds__(256) void gate_kernel(
    const float* __restrict__ x, const float* __restrict__ Wg,
    const float* __restrict__ bg, float* __restrict__ gates) {
    const int wave = threadIdx.x >> 5;
    const int lane = threadIdx.x & 31;
    const int row  = blockIdx.x * 8 + wave;
    const float* xr = x + (size_t)row * IDIM;

    float acc = bg[lane];                 // lane c = e*4 + t
#pragma unroll 8
    for (int i = 0; i < IDIM; ++i)
        acc = fmaf(xr[i], Wg[i * 32 + lane], acc);

    float m = acc;
    m = fmaxf(m, __shfl_xor(m, 4, 32));
    m = fmaxf(m, __shfl_xor(m, 8, 32));
    m = fmaxf(m, __shfl_xor(m, 16, 32));
    float p = __expf(acc - m);
    float s = p;
    s += __shfl_xor(s, 4, 32);
    s += __shfl_xor(s, 8, 32);
    s += __shfl_xor(s, 16, 32);
    const int t = lane & 3, e = lane >> 2;
    gates[(size_t)row * 32 + t * 8 + e] = p / s;
}

// ---- Kernel 4: expert GEMM (bf16 WMMA, fragments direct from global) --------
//                + relu + gated combine, expert tile kept in LDS
__global__ __launch_bounds__(256) void moe_kernel(
    const unsigned short* __restrict__ xb,   // bf16 [B][512]
    const unsigned short* __restrict__ wet,  // bf16 [2048][512]
    const float* __restrict__ be, const float* __restrict__ gates,
    float* __restrict__ out) {
    extern __shared__ char smem[];
    float* Gls = (float*)smem;                        // 16 KB
    float* Eo  = (float*)(smem + SMEM_GATE_BYTES);    // 66 KB

    const int tid    = threadIdx.x;
    const int wave   = tid >> 5;
    const int lane   = tid & 31;
    const int wave_m = wave & 3;
    const int wave_n = wave >> 2;
    const int row0   = blockIdx.x * BM;
    const int n0     = blockIdx.y * BN;
    const int u0     = blockIdx.y * (BN / NEXP);
    const int half   = lane >> 4;
    const int nl     = lane & 15;

    // preload gates for this row tile
#pragma unroll
    for (int q = 0; q < 4; ++q) {
        int off = q * 1024 + tid * 4;
        *(float4*)&Gls[off] = *(const float4*)&gates[(size_t)row0 * 32 + off];
    }

    float beR[4];
#pragma unroll
    for (int j = 0; j < 4; ++j)
        beR[j] = be[n0 + wave_n * 64 + j * 16 + nl];

    v8f acc[2][4];
#pragma unroll
    for (int i = 0; i < 2; ++i)
#pragma unroll
        for (int j = 0; j < 4; ++j)
            acc[i][j] = (v8f)0.0f;

    const int kbA = half * 8;    // A frag: chunks {kbA..kbA+7, kbA+16..kbA+23}
    const int kbB = half * 8;    // B frag: chunks {kbB..kbB+7, kbB+16..kbB+23}

    const unsigned short* pa = xb  + (size_t)(row0 + wave_m * 32 + nl) * IDIM;
    const unsigned short* pb = wet + (size_t)(n0  + wave_n * 64 + nl) * IDIM;

    for (int kk = 0; kk < IDIM; kk += BK) {
        __builtin_prefetch(pa + kk + BK, 0, 3);   // global_prefetch_b8
        __builtin_prefetch(pb + kk + BK, 0, 3);
        BF16x16 af[2], bf[4];
#pragma unroll
        for (int i = 0; i < 2; ++i) {
            const unsigned short* p = pa + (size_t)i * 16 * IDIM + kk + kbA;
            af[i].h[0] = *(const v8bf*)p;
            af[i].h[1] = *(const v8bf*)(p + 16);
        }
#pragma unroll
        for (int j = 0; j < 4; ++j) {
            const unsigned short* p = pb + (size_t)j * 16 * IDIM + kk + kbB;
            bf[j].h[0] = *(const v8bf*)p;
            bf[j].h[1] = *(const v8bf*)(p + 16);
        }
#pragma unroll
        for (int i = 0; i < 2; ++i)
#pragma unroll
            for (int j = 0; j < 4; ++j)
                acc[i][j] = __builtin_amdgcn_wmma_f32_16x16x32_bf16(
                    false, af[i].v, false, bf[j].v,
                    (short)0, acc[i][j], false, false);
    }

    // ---- relu(acc + be) -> LDS expert-output tile Eo[128][132] ----
#pragma unroll
    for (int i = 0; i < 2; ++i)
#pragma unroll
        for (int j = 0; j < 4; ++j) {
            int n = wave_n * 64 + j * 16 + nl;
#pragma unroll
            for (int v = 0; v < 8; ++v) {
                int m = wave_m * 32 + i * 16 + half * 8 + v;
                Eo[m * LDEO + n] = fmaxf(acc[i][j][v] + beR[j], 0.0f);
            }
        }
    __syncthreads();

    // ---- combine: out[t][b][u] = sum_e Eo[b][u*8+e] * g[b][t][e] ----
#pragma unroll
    for (int r = 0; r < 2; ++r) {
        int p = tid + r * 256;       // (t, b_local)
        int t = p >> 7;
        int bl = p & 127;
        float4 g0 = *(const float4*)&Gls[bl * 32 + t * 8];
        float4 g1 = *(const float4*)&Gls[bl * 32 + t * 8 + 4];
        float* outp = out + (((size_t)t * BATCH + (row0 + bl)) * UNITS + u0);
#pragma unroll
        for (int uq = 0; uq < 4; ++uq) {
            float4 res;
            float* resp = (float*)&res;
#pragma unroll
            for (int c = 0; c < 4; ++c) {
                int ul = uq * 4 + c;
                float4 a0 = *(const float4*)&Eo[bl * LDEO + ul * 8];
                float4 a1 = *(const float4*)&Eo[bl * LDEO + ul * 8 + 4];
                resp[c] = a0.x * g0.x + a0.y * g0.y + a0.z * g0.z + a0.w * g0.w
                        + a1.x * g1.x + a1.y * g1.y + a1.z * g1.z + a1.w * g1.w;
            }
            *(float4*)(outp + uq * 4) = res;
        }
    }
}

// ---- Launch -----------------------------------------------------------------
extern "C" void kernel_launch(void* const* d_in, const int* in_sizes, int n_in,
                              void* d_out, int out_size, void* d_ws, size_t ws_size,
                              hipStream_t stream) {
    const float* x  = (const float*)d_in[0];
    const float* We = (const float*)d_in[1];
    const float* be = (const float*)d_in[2];
    const float* Wg = (const float*)d_in[3];
    const float* bg = (const float*)d_in[4];
    float* out = (float*)d_out;

    float*          gates = (float*)((char*)d_ws + WS_GATES_OFF);
    unsigned short* xb    = (unsigned short*)((char*)d_ws + WS_XB_OFF);
    unsigned short* wet   = (unsigned short*)((char*)d_ws + WS_WET_OFF);

    convx_kernel<<<(BATCH * IDIM) / (256 * 8), 256, 0, stream>>>(x, xb);
    wet_kernel<<<dim3(IDIM / TT, NCOLS / TT), 256, 0, stream>>>(We, wet);
    gate_kernel<<<BATCH / 8, 256, 0, stream>>>(x, Wg, bg, gates);
    moe_kernel<<<dim3(BATCH / BM, NCOLS / BN), 256, SMEM_TOTAL, stream>>>(
        xb, wet, be, gates, out);
}
